// GraphFeatureFusion_88175678587174
// MI455X (gfx1250) — matrix operand order repaired and checked by reading, not compiled, not run on hardware
//
#include <hip/hip_runtime.h>
#include <hip/hip_bf16.h>
#include <stdint.h>

// Problem constants (match reference)
#define B_    16
#define NP_   1024
#define FEAT_ 512
#define H_    1024
#define E_    (B_ * NP_ * 16)   // 262144

typedef __bf16 v16bf __attribute__((ext_vector_type(16)));
typedef float  v8f   __attribute__((ext_vector_type(8)));

static __device__ __forceinline__ unsigned short f2bf(float f) {
  union { float f; unsigned u; } c; c.f = f;
  unsigned u = c.u;
  u += 0x7FFFu + ((u >> 16) & 1u);   // round-to-nearest-even
  return (unsigned short)(u >> 16);
}

// ---------------------------------------------------------------- utilities
__global__ void k_zero(float* p, long n) {
  long i = (long)blockIdx.x * blockDim.x + threadIdx.x;
  if (i < n) p[i] = 0.0f;
}

__global__ void k_edge_init(const int* ei, int* cs, int* cd, int e) {
  int i = blockIdx.x * blockDim.x + threadIdx.x;
  if (i < e) { cs[i] = ei[i]; cd[i] = ei[E_ + i]; }
}

// Concatenate Wl|Wr along K and convert to bf16: Wcat[n][k], k in [0,2F)
__global__ void k_wcat(const float* Wl, const float* Wr, unsigned short* Wcat, int F) {
  long i = (long)blockIdx.x * blockDim.x + threadIdx.x;
  long total = (long)H_ * 2 * F;
  if (i >= total) return;
  int n = (int)(i / (2 * F));
  int k = (int)(i % (2 * F));
  float v = (k < F) ? Wl[(long)n * F + k] : Wr[(long)n * F + (k - F)];
  Wcat[i] = f2bf(v);
}

// ---------------------------------------------------------------- scatter (mean agg numerator + degree)
__global__ void k_scatter(const int* cs, const int* cd, const float* xin,
                          float* agg, float* deg, int F) {
  long i = (long)blockIdx.x * blockDim.x + threadIdx.x;
  long total = (long)E_ * (F >> 2);
  if (i >= total) return;
  int e = (int)(i / (F >> 2));
  int c = (int)(i % (F >> 2)) * 4;
  int s = cs[e], d = cd[e];
  if (s < 0 || d < 0) return;
  const float* xr = xin + (long)s * F + c;
  float*       ar = agg + (long)d * F + c;
  atomicAdd(ar + 0, xr[0]);
  atomicAdd(ar + 1, xr[1]);
  atomicAdd(ar + 2, xr[2]);
  atomicAdd(ar + 3, xr[3]);
  if (c == 0) atomicAdd(deg + d, 1.0f);
}

// ---------------------------------------------------------------- fused SAGE GEMM (WMMA bf16)
// out[m][n] = relu( bias[n] + sum_{k<F} (agg[m][k]/max(deg,1)) * Wcat[n][k]
//                            + sum_{F<=k<2F} xin[m][k-F] * Wcat[n][k] )
// Block tile 64(M) x 256(N); 8 waves as 2(M) x 4(N); wave tile 32x64; K-chunk 32.
// B tile staged via CDNA5 async global->LDS DMA (ASYNCcnt).
__global__ __launch_bounds__(256) void k_sage_gemm(
    const float* __restrict__ agg, const float* __restrict__ deg,
    const float* __restrict__ xin, const unsigned short* __restrict__ Wcat,
    const float* __restrict__ bias, float* __restrict__ out, int F, int Nl) {
  const int K2 = 2 * F;
  __shared__ alignas(16) unsigned short As[64][32];     // [m][k] bf16    (4 KB)
  __shared__ alignas(16) unsigned short Bs[256][32];    // [n][k] bf16    (16 KB, N-major)

  const int tid   = threadIdx.x;
  const int nBase = blockIdx.x * 256;
  const int mBase = blockIdx.y * 64;
  const int w     = tid >> 5;
  const int lane  = tid & 31;
  const int g     = lane >> 4;     // half-wave group
  const int ln    = lane & 15;
  const int waveM = w >> 2;        // 0..1
  const int waveN = w & 3;         // 0..3

  // A tile loader mapping: 256 threads cover 64 rows x 32 cols (8 elems each)
  const int ar = tid >> 2;             // 0..63
  const int ac = (tid & 3) * 8;        // 0,8,16,24
  const int arow = mBase + ar;
  const float rd = 1.0f / fmaxf(deg[arow], 1.0f);

  // B tile loader: thread t DMAs one full 64-byte row (32 bf16) of Bs
  const unsigned ldsB = (unsigned)(unsigned long long)&Bs[tid][0];

  v8f acc[2][4] = {};

  #pragma unroll
  for (int phase = 0; phase < 2; ++phase) {
    const float* __restrict__ asrc = phase ? xin : agg;   // mean numerator then raw x
    const float  ascale            = phase ? 1.0f : rd;
    const int    kofs              = phase ? F : 0;

    for (int kb = 0; kb < F; kb += 32) {
      __syncthreads();
      // ---- stage A (fp32 -> bf16, scaled)
      {
        const float* srcA = asrc + (long)arow * F + kb + ac;
        float4 a0 = *(const float4*)(srcA);
        float4 a1 = *(const float4*)(srcA + 4);
        if (kb + 32 < F)
          __builtin_prefetch(srcA + 32, 0, 1);   // global_prefetch_b8 next K-chunk
        uint4 p;
        p.x = (unsigned)f2bf(a0.x * ascale) | ((unsigned)f2bf(a0.y * ascale) << 16);
        p.y = (unsigned)f2bf(a0.z * ascale) | ((unsigned)f2bf(a0.w * ascale) << 16);
        p.z = (unsigned)f2bf(a1.x * ascale) | ((unsigned)f2bf(a1.y * ascale) << 16);
        p.w = (unsigned)f2bf(a1.z * ascale) | ((unsigned)f2bf(a1.w * ascale) << 16);
        *(uint4*)&As[ar][ac] = p;
      }
      // ---- stage B via async DMA: 64B row, INST_OFFSET applies to both LDS+global
      {
        const unsigned short* srcB = Wcat + (long)(nBase + tid) * K2 + kofs + kb;
        asm volatile(
            "global_load_async_to_lds_b128 %0, %1, off\n\t"
            "global_load_async_to_lds_b128 %0, %1, off offset:16\n\t"
            "global_load_async_to_lds_b128 %0, %1, off offset:32\n\t"
            "global_load_async_to_lds_b128 %0, %1, off offset:48"
            :: "v"(ldsB), "v"(srcB) : "memory");
        asm volatile("s_wait_asynccnt 0x0" ::: "memory");
      }
      __syncthreads();

      // ---- fragments from LDS (ISA 16-bit A 16x32 / B N-major layouts)
      union { uint4 u[2]; v16bf v; } fa[2], fb[4];
      #pragma unroll
      for (int mt = 0; mt < 2; ++mt) {
        int r = waveM * 32 + mt * 16 + ln;
        fa[mt].u[0] = *(const uint4*)&As[r][8 * g];        // K = 8g .. 8g+7
        fa[mt].u[1] = *(const uint4*)&As[r][16 + 8 * g];   // K = 16+8g .. 16+8g+7
      }
      #pragma unroll
      for (int nt = 0; nt < 4; ++nt) {
        int r = waveN * 64 + nt * 16 + ln;
        fb[nt].u[0] = *(const uint4*)&Bs[r][8 * g];        // K = 8g .. 8g+7
        fb[nt].u[1] = *(const uint4*)&Bs[r][16 + 8 * g];   // K = 16+8g .. 16+8g+7
      }
      #pragma unroll
      for (int mt = 0; mt < 2; ++mt)
        #pragma unroll
        for (int nt = 0; nt < 4; ++nt)
          acc[mt][nt] = __builtin_amdgcn_wmma_f32_16x16x32_bf16(
              false, fa[mt].v, false, fb[nt].v, (short)0, acc[mt][nt], false, false);
    }
  }

  // ---- epilogue: bias + relu (C layout: M = r + 8g, N = ln)
  #pragma unroll
  for (int mt = 0; mt < 2; ++mt) {
    #pragma unroll
    for (int nt = 0; nt < 4; ++nt) {
      int N = nBase + waveN * 64 + nt * 16 + ln;
      float bv = bias[N];
      #pragma unroll
      for (int r = 0; r < 8; ++r) {
        int M = mBase + waveM * 32 + mt * 16 + 8 * g + r;
        float v = acc[mt][nt][r] + bv;
        out[(long)M * H_ + N] = v > 0.0f ? v : 0.0f;
      }
    }
  }
}

// ---------------------------------------------------------------- topk scoring
__global__ void k_pnorm(const float* p, float* pn) {
  __shared__ float red[256];
  float s = 0.0f;
  for (int i = threadIdx.x; i < H_; i += 256) { float v = p[i]; s += v * v; }
  red[threadIdx.x] = s; __syncthreads();
  for (int o = 128; o > 0; o >>= 1) {
    if (threadIdx.x < o) red[threadIdx.x] += red[threadIdx.x + o];
    __syncthreads();
  }
  if (threadIdx.x == 0) pn[0] = sqrtf(red[0]);
}

__global__ void k_score(const float* h, const float* p, const float* pn, float* s) {
  __shared__ float red[256];
  const int node = blockIdx.x;
  const float* hr = h + (long)node * H_;
  float acc = 0.0f;
  for (int i = threadIdx.x; i < H_; i += 256) acc += hr[i] * p[i];
  red[threadIdx.x] = acc; __syncthreads();
  for (int o = 128; o > 0; o >>= 1) {
    if (threadIdx.x < o) red[threadIdx.x] += red[threadIdx.x + o];
    __syncthreads();
  }
  if (threadIdx.x == 0) s[node] = tanhf(red[0] / pn[0]);
}

// Per-graph top-k via LDS bitonic sort (descending), 1024 slots padded with -inf.
__global__ __launch_bounds__(512) void k_topk(const float* s, int n_per, int k,
                                              float* vals, int* oldidx, int* remap) {
  __shared__ float v[1024];
  __shared__ int   ix[1024];
  const int g = blockIdx.x, tid = threadIdx.x;
  for (int i = tid; i < 1024; i += 512) {
    v[i]  = (i < n_per) ? s[(long)g * n_per + i] : -1e30f;
    ix[i] = i;
  }
  for (int i = tid; i < n_per; i += 512) remap[(long)g * n_per + i] = -1;
  __syncthreads();
  for (int ksz = 2; ksz <= 1024; ksz <<= 1) {
    for (int j = ksz >> 1; j > 0; j >>= 1) {
      for (int i = tid; i < 1024; i += 512) {
        int l = i ^ j;
        if (l > i) {
          bool sw = ((i & ksz) == 0) ? (v[i] < v[l]) : (v[i] > v[l]);
          if (sw) {
            float tv = v[i]; v[i] = v[l]; v[l] = tv;
            int   ti = ix[i]; ix[i] = ix[l]; ix[l] = ti;
          }
        }
      }
      __syncthreads();
    }
  }
  for (int jj = tid; jj < k; jj += 512) {
    int og  = g * n_per + ix[jj];
    int nid = g * k + jj;
    vals[nid]   = v[jj];
    oldidx[nid] = og;
    remap[og]   = nid;
  }
}

// Gather kept nodes, gated by score
__global__ void k_pool(const float* h, const float* vals, const int* oldidx,
                       float* hp, long total) {
  long i = (long)blockIdx.x * blockDim.x + threadIdx.x;
  if (i >= total) return;
  int nid = (int)(i / H_);
  int f   = (int)(i % H_);
  hp[i] = h[(long)oldidx[nid] * H_ + f] * vals[nid];
}

__global__ void k_remap(int* cs, int* cd, const int* remap, int e) {
  int i = blockIdx.x * blockDim.x + threadIdx.x;
  if (i >= e) return;
  int s = cs[i], d = cd[i];
  int ns = -1, nd = -1;
  if (s >= 0 && d >= 0) { ns = remap[s]; nd = remap[d]; }
  if (ns < 0 || nd < 0) { cs[i] = -1; cd[i] = -1; }
  else                  { cs[i] = ns; cd[i] = nd; }
}

// out[b][0:H] += max over kept nodes; out[b][H:2H] += mean over k
__global__ void k_readout(const float* hp, float* out, int k) {
  int i = blockIdx.x * blockDim.x + threadIdx.x;
  if (i >= B_ * H_) return;
  int g = i / H_, f = i % H_;
  const float* base = hp + ((long)g * k) * H_ + f;
  float mx = -1e30f, sum = 0.0f;
  for (int j = 0; j < k; ++j) {
    float v = base[(long)j * H_];
    mx = fmaxf(mx, v);
    sum += v;
  }
  out[(long)g * 2 * H_ + f]      += mx;
  out[(long)g * 2 * H_ + H_ + f] += sum / (float)k;
}

// ---------------------------------------------------------------- orchestration
extern "C" void kernel_launch(void* const* d_in, const int* in_sizes, int n_in,
                              void* d_out, int out_size, void* d_ws, size_t ws_size,
                              hipStream_t stream) {
  const float* x  = (const float*)d_in[0];
  const int*   ei = (const int*)d_in[1];
  const float* Wl[3] = {(const float*)d_in[3], (const float*)d_in[7],  (const float*)d_in[11]};
  const float* bb[3] = {(const float*)d_in[4], (const float*)d_in[8],  (const float*)d_in[12]};
  const float* Wr[3] = {(const float*)d_in[5], (const float*)d_in[9],  (const float*)d_in[13]};
  const float* pp[3] = {(const float*)d_in[6], (const float*)d_in[10], (const float*)d_in[14]};
  float* out = (float*)d_out;

  char* ws = (char*)d_ws;
  size_t off = 0;
  auto alloc = [&](size_t bytes) -> char* {
    char* p = ws + off;
    off = (off + bytes + 255) & ~(size_t)255;
    return p;
  };
  float*          h      = (float*)alloc((size_t)16384 * 1024 * 4);
  float*          agg    = (float*)alloc((size_t)16384 * 1024 * 4);
  float*          hp     = (float*)alloc((size_t)13120 * 1024 * 4);
  unsigned short* Wcat   = (unsigned short*)alloc((size_t)1024 * 2048 * 2);
  float*          deg    = (float*)alloc(16384 * 4);
  float*          sc     = (float*)alloc(16384 * 4);
  float*          vals   = (float*)alloc(16384 * 4);
  int*            oldidx = (int*)alloc(16384 * 4);
  int*            remap  = (int*)alloc(16384 * 4);
  int*            cs     = (int*)alloc((size_t)E_ * 4);
  int*            cd     = (int*)alloc((size_t)E_ * 4);
  float*          pn     = (float*)alloc(256);

  k_zero<<<(B_ * 2 * H_ + 255) / 256, 256, 0, stream>>>(out, B_ * 2 * H_);
  k_edge_init<<<(E_ + 255) / 256, 256, 0, stream>>>(ei, cs, cd, E_);

  int n_per = NP_;
  int F = FEAT_;
  const float* xin = x;

  for (int layer = 0; layer < 3; ++layer) {
    const int Nl = B_ * n_per;
    const int k  = (4 * n_per + 4) / 5;     // ceil(0.8*n): 820, 656, 525

    long wtot = (long)H_ * 2 * F;
    k_wcat<<<(unsigned)((wtot + 255) / 256), 256, 0, stream>>>(Wl[layer], Wr[layer], Wcat, F);

    k_zero<<<(unsigned)(((long)Nl * F + 255) / 256), 256, 0, stream>>>(agg, (long)Nl * F);
    k_zero<<<(Nl + 255) / 256, 256, 0, stream>>>(deg, Nl);

    long st = (long)E_ * (F / 4);
    k_scatter<<<(unsigned)((st + 255) / 256), 256, 0, stream>>>(cs, cd, xin, agg, deg, F);

    dim3 gg(H_ / 256, Nl / 64);
    k_sage_gemm<<<gg, 256, 0, stream>>>(agg, deg, xin, Wcat, bb[layer], h, F, Nl);

    k_pnorm<<<1, 256, 0, stream>>>(pp[layer], pn);
    k_score<<<Nl, 256, 0, stream>>>(h, pp[layer], pn, sc);

    k_topk<<<B_, 512, 0, stream>>>(sc, n_per, k, vals, oldidx, remap);

    long pt = (long)B_ * k * H_;
    k_pool<<<(unsigned)((pt + 255) / 256), 256, 0, stream>>>(h, vals, oldidx, hp, pt);

    k_remap<<<(E_ + 255) / 256, 256, 0, stream>>>(cs, cd, remap, E_);

    k_readout<<<(B_ * H_ + 255) / 256, 256, 0, stream>>>(hp, out, k);

    xin   = hp;
    F     = H_;
    n_per = k;
  }
}
